// TimeLagTransitionDecoder_9354438771318
// MI455X (gfx1250) — matrix operand order repaired and checked by reading, not compiled, not run on hardware
//
#include <hip/hip_runtime.h>
#include <hip/hip_bf16.h>

typedef __attribute__((ext_vector_type(16))) _Float16 v16h;
typedef __attribute__((ext_vector_type(8)))  float    v8f;

#define D_X 16
#define D_L 8
#define T_LEN 131072
#define BIN_THRES 0.1f
#define TTILE 16

// Output layout (floats), in reference return order:
//   recon [T,16]      @ 0
//   pred  [T,16]      @ 16T
//   lm_pred [2,T,8]   @ 32T
//   lv_pred [2,T,8]   @ 48T
//   Ct    [T,16,8]    @ 64T
//   Ct_1  [T,16,8]    @ 192T
#define OFF_RECON ((size_t)0)
#define OFF_PRED  ((size_t)16 * T_LEN)
#define OFF_LM    ((size_t)32 * T_LEN)
#define OFF_LV    ((size_t)48 * T_LEN)
#define OFF_CT    ((size_t)64 * T_LEN)
#define OFF_CT1   ((size_t)192 * T_LEN)

__device__ __forceinline__ float lrelu(float x) { return x > 0.f ? x : 0.01f * x; }

__device__ __forceinline__ unsigned pack2h(float a, float b) {
    _Float16 ha = (_Float16)a, hb = (_Float16)b;
    unsigned short ua, ub;
    __builtin_memcpy(&ua, &ha, 2);
    __builtin_memcpy(&ub, &hb, 2);
    return (unsigned)ua | ((unsigned)ub << 16);
}

union FragU {
    uint4 q[2];
    v16h  h;
};

// ---------------------------------------------------------------------------
// Prep: convert lw0 (16,32,64) and lw1 (16,64,64) f32 weights into per-lane
// WMMA B-fragments (f16, packed 2/dword) in workspace.
//   W0 frag (n,j4):     dwords [ (n*4+j4)*256 + lane*8 .. +7 ]
//   W1 frag (n,kc,j4):  dwords [ 16384 + (n*8+kc*4+j4)*256 + lane*8 .. +7 ]
// B layout: lane&15 = column N within 16-wide tile; lanes>=16 cover K=16..31;
// dword j holds (K=kb+2j, K=kb+2j+1).
// ---------------------------------------------------------------------------
__global__ void __launch_bounds__(32)
prep_frags_kernel(const float* __restrict__ lw0, const float* __restrict__ lw1,
                  unsigned* __restrict__ frag) {
    int b = blockIdx.x;
    int lane = threadIdx.x;
    int col = lane & 15;
    int kb = (lane >> 4) * 16;
    if (b < 64) {                       // layer-0 fragments
        int n = b >> 2, j4 = b & 3;
        unsigned* dst = frag + (size_t)b * 256 + lane * 8;
        #pragma unroll
        for (int j = 0; j < 8; ++j) {
            int k = kb + 2 * j;
            float v0 = lw0[(n * 32 + k    ) * 64 + j4 * 16 + col];
            float v1 = lw0[(n * 32 + k + 1) * 64 + j4 * 16 + col];
            dst[j] = pack2h(v0, v1);
        }
    } else {                            // layer-1 fragments
        int idx = b - 64;               // idx = n*8 + kc*4 + j4
        int n = idx >> 3, kc = (idx >> 2) & 1, j4 = idx & 3;
        unsigned* dst = frag + (size_t)16384 + (size_t)idx * 256 + lane * 8;
        #pragma unroll
        for (int j = 0; j < 8; ++j) {
            int k = kb + 2 * j;
            float v0 = lw1[(n * 64 + kc * 32 + k    ) * 64 + j4 * 16 + col];
            float v1 = lw1[(n * 64 + kc * 32 + k + 1) * 64 + j4 * 16 + col];
            dst[j] = pack2h(v0, v1);
        }
    }
}

// ---------------------------------------------------------------------------
// Heads: lm_pred / lv_pred = relu(x @ w1 + b1) @ w2 + b2, per (b,t).
// ---------------------------------------------------------------------------
__global__ void __launch_bounds__(256)
heads_kernel(const float* __restrict__ lm, const float* __restrict__ lv,
             const float* __restrict__ fmw1, const float* __restrict__ fmb1,
             const float* __restrict__ fmw2, const float* __restrict__ fmb2,
             const float* __restrict__ fvw1, const float* __restrict__ fvb1,
             const float* __restrict__ fvw2, const float* __restrict__ fvb2,
             float* __restrict__ out) {
    __shared__ float sM1[1024], sM2[1024], sMb[128];
    __shared__ float sV1[1024], sV2[1024], sVb[128];
    int tid = threadIdx.x;
    for (int i = tid; i < 1024; i += 256) {
        sM1[i] = fmw1[i]; sM2[i] = fmw2[i];
        sV1[i] = fvw1[i]; sV2[i] = fvw2[i];
    }
    if (tid < 128) { sMb[tid] = fmb1[tid]; sVb[tid] = fvb1[tid]; }
    __syncthreads();

    size_t gid = (size_t)blockIdx.x * 256 + tid;      // (b,t), gid < 2*T
    size_t base = gid * 8;

    // lm head
    {
        float x[8], o[8];
        #pragma unroll
        for (int k = 0; k < 8; ++k) { x[k] = lm[base + k]; o[k] = fmb2[k]; }
        for (int j = 0; j < 128; ++j) {
            float s = sMb[j];
            #pragma unroll
            for (int i = 0; i < 8; ++i) s += x[i] * sM1[i * 128 + j];
            s = fmaxf(s, 0.f);
            #pragma unroll
            for (int k = 0; k < 8; ++k) o[k] += s * sM2[j * 8 + k];
        }
        #pragma unroll
        for (int k = 0; k < 8; ++k) out[OFF_LM + base + k] = o[k];
    }
    // lv head
    {
        float x[8], o[8];
        #pragma unroll
        for (int k = 0; k < 8; ++k) { x[k] = lv[base + k]; o[k] = fvb2[k]; }
        for (int j = 0; j < 128; ++j) {
            float s = sVb[j];
            #pragma unroll
            for (int i = 0; i < 8; ++i) s += x[i] * sV1[i * 128 + j];
            s = fmaxf(s, 0.f);
            #pragma unroll
            for (int k = 0; k < 8; ++k) o[k] += s * sV2[j * 8 + k];
        }
        #pragma unroll
        for (int k = 0; k < 8; ++k) out[OFF_LV + base + k] = o[k];
    }
}

// ---------------------------------------------------------------------------
// Decode: one block = 16 timesteps. Stage 1 stages f16 x/C into LDS and also
// emits Ct/Ct_1 f32 outputs. Stage 2: 8 waves x 4 (decode,node) WMMA tasks.
// ---------------------------------------------------------------------------
__global__ void __launch_bounds__(256)
decode_kernel(const float* __restrict__ lm, const float* __restrict__ lv,
              const float* __restrict__ Tin,
              const float* __restrict__ cw0, const float* __restrict__ cb0,
              const float* __restrict__ cw1, const float* __restrict__ cb1,
              const float* __restrict__ cw2, const float* __restrict__ cb2,
              const float* __restrict__ pw0, const float* __restrict__ pb0,
              const float* __restrict__ pw1, const float* __restrict__ pb1,
              const float* __restrict__ pw2, const float* __restrict__ pb2,
              const float* __restrict__ lw2, const float* __restrict__ lb0,
              const float* __restrict__ lb1, const float* __restrict__ lb2,
              const unsigned* __restrict__ frag,
              float* __restrict__ out) {
    __shared__ _Float16 sX [4][TTILE][8];      // [lm0,lm1,lv0,lv1] current
    __shared__ _Float16 sXp[4][TTILE][8];      // shifted (prev) versions
    __shared__ _Float16 sC [2][TTILE][D_X][8]; // [0]=Ct(bin), [1]=Ct_1
    __shared__ float sW2[1024];                // lw2 (16,64)
    __shared__ float sB0[1024];                // lb0 (16,64)
    __shared__ float sB1[1024];                // lb1 (16,64)
    __shared__ float sB2[16];                  // lb2 (16)
    __shared__ float sH[8][TTILE][66];         // per-wave 16x64 f32 scratch (+pad)

    const int tid = threadIdx.x;
    const int t0 = blockIdx.x * TTILE;

    // ---- Stage 1: cooperative staging -------------------------------------
    for (int i = tid; i < 1024; i += 256) {
        sW2[i] = lw2[i]; sB0[i] = lb0[i]; sB1[i] = lb1[i];
    }
    if (tid < 16) sB2[tid] = lb2[tid];

    for (int idx = tid; idx < 512; idx += 256) {
        int a  = idx >> 7;          // 0..3
        int tl = (idx >> 3) & 15;
        int k  = idx & 7;
        int b  = a & 1;
        const float* src = (a < 2) ? lm : lv;
        int t = t0 + tl;
        sX[a][tl][k] = (_Float16)src[((size_t)b * T_LEN + t) * 8 + k];
        float pv = (t == 0) ? 0.f : src[((size_t)b * T_LEN + (t - 1)) * 8 + k];
        sXp[a][tl][k] = (_Float16)pv;
    }

    if (tid < 32) {                  // Ct / Ct_1 tiny MLPs (1->8->8->128)
        int which = tid >> 4;        // 0: curr (binarized), 1: pre
        int tl = tid & 15;
        int t  = t0 + tl;
        const float* w0 = which ? pw0 : cw0; const float* b0 = which ? pb0 : cb0;
        const float* w1 = which ? pw1 : cw1; const float* b1 = which ? pb1 : cb1;
        const float* w2 = which ? pw2 : cw2; const float* b2 = which ? pb2 : cb2;
        float tv = Tin[t];
        float h0[8], h1[8];
        #pragma unroll
        for (int j = 0; j < 8; ++j) h0[j] = lrelu(tv * w0[j] + b0[j]);
        #pragma unroll
        for (int j = 0; j < 8; ++j) {
            float s = b1[j];
            #pragma unroll
            for (int i = 0; i < 8; ++i) s += h0[i] * w1[i * 8 + j];
            h1[j] = lrelu(s);
        }
        for (int o = 0; o < 128; ++o) {
            float s = b2[o];
            #pragma unroll
            for (int i = 0; i < 8; ++i) s += h1[i] * w2[i * 128 + o];
            if (which == 0) {
                float bin = (s < BIN_THRES) ? 0.f : 1.f;
                out[OFF_CT + (size_t)t * 128 + o] = bin;
                sC[0][tl][o >> 3][o & 7] = (_Float16)bin;
            } else {
                out[OFF_CT1 + (size_t)t * 128 + o] = s;
                sC[1][tl][o >> 3][o & 7] = (_Float16)s;
            }
        }
    }
    __syncthreads();

    // ---- Stage 2: WMMA tasks ----------------------------------------------
    const int w    = tid >> 5;
    const int lane = tid & 31;
    const int m    = lane & 15;      // row within tile (A) / col (D)
    const int hi   = lane >> 4;
    const uint4* wsq = (const uint4*)frag;

    for (int it = 0; it < 4; ++it) {
        int task = w + 8 * it;       // 0..31
        int n = task & 15;           // node
        int d = task >> 4;           // 0: recon (Ct, curr), 1: pred (Ct_1, prev)

        // Build A (16x32 f16): features f -> C[t,n,f>>1 (mod 8)] * x
        v16h a;
        #pragma unroll
        for (int v = 0; v < 8; ++v) {
            int k0 = 16 * hi + 2 * v;
            #pragma unroll
            for (int e = 0; e < 2; ++e) {
                int f  = k0 + e;
                int wh = f >> 4;             // 0: lm, 1: lv
                int li = f & 1;              // batch index
                int kk = (f >> 1) & 7;       // latent dim
                _Float16 xv = d ? sXp[wh * 2 + li][m][kk] : sX[wh * 2 + li][m][kk];
                _Float16 cv = sC[d][m][n][kk];
                a[2 * v + e] = xv * cv;
            }
        }

        // Layer 0: 4 WMMAs, bias + relu, spill H1 to LDS
        #pragma unroll
        for (int j4 = 0; j4 < 4; ++j4) {
            FragU bf;
            size_t off = ((size_t)(n * 4 + j4)) * 64 + lane * 2;
            bf.q[0] = wsq[off]; bf.q[1] = wsq[off + 1];
            v8f c = {0.f, 0.f, 0.f, 0.f, 0.f, 0.f, 0.f, 0.f};
            c = __builtin_amdgcn_wmma_f32_16x16x32_f16(false, a, false, bf.h,
                                                       (short)0, c, false, false);
            float bias = sB0[n * 64 + j4 * 16 + m];
            #pragma unroll
            for (int r = 0; r < 8; ++r)
                sH[w][r + 8 * hi][j4 * 16 + m] = fmaxf(c[r] + bias, 0.f);
        }
        __syncthreads();

        // Layer 1: rebuild A fragments from LDS, 8 WMMAs, bias (no relu)
        v16h a2[2];
        #pragma unroll
        for (int kc = 0; kc < 2; ++kc) {
            #pragma unroll
            for (int v = 0; v < 8; ++v) {
                int k0 = 16 * hi + 2 * v;
                a2[kc][2 * v    ] = (_Float16)sH[w][m][kc * 32 + k0];
                a2[kc][2 * v + 1] = (_Float16)sH[w][m][kc * 32 + k0 + 1];
            }
        }
        float h2b[4][8];
        #pragma unroll
        for (int j4 = 0; j4 < 4; ++j4) {
            v8f c2 = {0.f, 0.f, 0.f, 0.f, 0.f, 0.f, 0.f, 0.f};
            #pragma unroll
            for (int kc = 0; kc < 2; ++kc) {
                FragU bf;
                size_t off = 4096 + ((size_t)(n * 8 + kc * 4 + j4)) * 64 + lane * 2;
                bf.q[0] = wsq[off]; bf.q[1] = wsq[off + 1];
                c2 = __builtin_amdgcn_wmma_f32_16x16x32_f16(false, a2[kc], false,
                                                            bf.h, (short)0, c2,
                                                            false, false);
            }
            float bias = sB1[n * 64 + j4 * 16 + m];
            #pragma unroll
            for (int r = 0; r < 8; ++r) h2b[j4][r] = c2[r] + bias;
        }
        // spill H2
        #pragma unroll
        for (int j4 = 0; j4 < 4; ++j4)
            #pragma unroll
            for (int r = 0; r < 8; ++r)
                sH[w][r + 8 * hi][j4 * 16 + m] = h2b[j4][r];
        __syncthreads();

        // Layer 2: 64 -> 1 dot + relu, lanes 0..15 write one row each
        float acc = sB2[n];
        #pragma unroll
        for (int c = 0; c < 64; ++c) acc += sH[w][m][c] * sW2[n * 64 + c];
        float o = fmaxf(acc, 0.f);
        if (lane < 16) {
            size_t base = d ? OFF_PRED : OFF_RECON;
            out[base + (size_t)(t0 + m) * D_X + n] = o;
        }
    }
}

// ---------------------------------------------------------------------------
extern "C" void kernel_launch(void* const* d_in, const int* in_sizes, int n_in,
                              void* d_out, int out_size, void* d_ws, size_t ws_size,
                              hipStream_t stream) {
    const float* lm   = (const float*)d_in[0];
    const float* lv   = (const float*)d_in[1];
    const float* Tin  = (const float*)d_in[2];
    const float* cw0  = (const float*)d_in[3];
    const float* cb0  = (const float*)d_in[4];
    const float* cw1  = (const float*)d_in[5];
    const float* cb1  = (const float*)d_in[6];
    const float* cw2  = (const float*)d_in[7];
    const float* cb2  = (const float*)d_in[8];
    const float* pw0  = (const float*)d_in[9];
    const float* pb0  = (const float*)d_in[10];
    const float* pw1  = (const float*)d_in[11];
    const float* pb1  = (const float*)d_in[12];
    const float* pw2  = (const float*)d_in[13];
    const float* pb2  = (const float*)d_in[14];
    const float* fmw1 = (const float*)d_in[15];
    const float* fmb1 = (const float*)d_in[16];
    const float* fmw2 = (const float*)d_in[17];
    const float* fmb2 = (const float*)d_in[18];
    const float* fvw1 = (const float*)d_in[19];
    const float* fvb1 = (const float*)d_in[20];
    const float* fvw2 = (const float*)d_in[21];
    const float* fvb2 = (const float*)d_in[22];
    const float* lw0  = (const float*)d_in[23];
    const float* lb0  = (const float*)d_in[24];
    const float* lw1  = (const float*)d_in[25];
    const float* lb1  = (const float*)d_in[26];
    const float* lw2  = (const float*)d_in[27];
    const float* lb2  = (const float*)d_in[28];
    float* out = (float*)d_out;
    unsigned* frag = (unsigned*)d_ws;     // needs 192 KB

    prep_frags_kernel<<<192, 32, 0, stream>>>(lw0, lw1, frag);
    heads_kernel<<<(2 * T_LEN) / 256, 256, 0, stream>>>(
        lm, lv, fmw1, fmb1, fmw2, fmb2, fvw1, fvb1, fvw2, fvb2, out);
    decode_kernel<<<T_LEN / TTILE, 256, 0, stream>>>(
        lm, lv, Tin, cw0, cb0, cw1, cb1, cw2, cb2,
        pw0, pb0, pw1, pb1, pw2, pb2,
        lw2, lb0, lb1, lb2, (const unsigned*)frag, out);
}
